// GNNActor_70798240907496
// MI455X (gfx1250) — compile-verified, matrix operand order checked by reading, not run on hardware
//
#include <hip/hip_runtime.h>
#include <math.h>

typedef __attribute__((ext_vector_type(16))) _Float16 v16h;
typedef __attribute__((ext_vector_type(8)))  _Float16 v8h;
typedef __attribute__((ext_vector_type(8)))  float    v8f;

#define N_NODES 60000
#define N_EDGES 240000
#define G_NUM   10000

// ---------------------------------------------------------------------------
// helpers
// ---------------------------------------------------------------------------
__device__ inline float waveSum(float v) {
#pragma unroll
    for (int off = 16; off > 0; off >>= 1) v += __shfl_xor(v, off, 32);
    return v;
}

__device__ inline void atomicMaxF(float* addr, float val) {
    unsigned int* ai = (unsigned int*)addr;
    unsigned int old = *ai;
    while (true) {
        float f = __uint_as_float(old);
        if (f >= val) break;
        unsigned int assumed = old;
        old = atomicCAS(ai, assumed, __float_as_uint(val));
        if (old == assumed) break;
    }
}

__device__ inline v16h combine8(v8h lo, v8h hi) {
    v16h f;
#pragma unroll
    for (int i = 0; i < 8; ++i) { f[i] = lo[i]; f[i + 8] = hi[i]; }
    return f;
}

// ---------------------------------------------------------------------------
// WMMA GEMM on pre-converted f16 operands.
//   C[M,N] (= | +=) A[M,K](f16) * W[N,K]^T(f16)  (+ bias)
// block = 128 threads (4 waves). Wave tile = 16 x (NT*16).
// B panel (NT*16 cols x K) is staged in LDS once per block, shared by all 4
// waves; rows are skewed by 8 halves so ds_load_b128 is bank-conflict free.
// GUARD=false requires N == gridDim.x * NT * 16.
// ---------------------------------------------------------------------------
template <int NT, bool GUARD>
__global__ __launch_bounds__(128)
void gemm_h(const _Float16* __restrict__ A, int lda,
            const _Float16* __restrict__ W, int ldw,
            const float* __restrict__ bias,
            float* __restrict__ C, int ldc,
            int M, int N, int K, int accumulate) {
    extern __shared__ _Float16 sh[];          // (NT*16) * (K+8) halves
    const int NCOLS = NT * 16;
    const int ld2 = K + 8;                    // skewed LDS row stride
    int n_base = blockIdx.x * NCOLS;

    // cooperative B-panel load: global -> LDS (v8h granules)
    int kdw = K >> 3;                         // K/8 granules per row
    for (int idx = threadIdx.x; idx < NCOLS * kdw; idx += 128) {
        int c = idx / kdw;
        int kk = (idx - c * kdw) * 8;
        v8h val = {};
        int wrow = n_base + c;
        if (!GUARD || wrow < N) val = *(const v8h*)(W + (size_t)wrow * ldw + kk);
        *(v8h*)(sh + c * ld2 + kk) = val;
    }
    __syncthreads();

    int lane = threadIdx.x & 31;
    int wave = threadIdx.x >> 5;
    int m0 = (blockIdx.y * 4 + wave) * 16;
    if (m0 >= M) return;                      // after the only barrier: safe

    v8f acc[NT] = {};
    int rowA = m0 + (lane & 15);
    if (rowA >= M) rowA = M - 1;              // clamped load; store is guarded
    int kh = (lane >> 4) * 8;

    for (int k0 = 0; k0 < K; k0 += 32) {
        const _Float16* ap = A + (size_t)rowA * lda + k0 + kh;
        v16h af = combine8(*(const v8h*)ap, *(const v8h*)(ap + 16));
#pragma unroll
        for (int nt = 0; nt < NT; ++nt) {
            const _Float16* bp = sh + (nt * 16 + (lane & 15)) * ld2 + k0 + kh;
            v16h bf = combine8(*(const v8h*)bp, *(const v8h*)(bp + 16));
            acc[nt] = __builtin_amdgcn_wmma_f32_16x16x32_f16(
                false, af, false, bf, (short)0, acc[nt], false, false);
        }
    }

    int colLane = lane & 15;
    int rhalf = (lane >> 4) * 8;
#pragma unroll
    for (int nt = 0; nt < NT; ++nt) {
        int n = n_base + nt * 16 + colLane;
        if (GUARD && n >= N) continue;
        float b = (!accumulate && bias) ? bias[n] : 0.0f;
#pragma unroll
        for (int r = 0; r < 8; ++r) {
            int row = m0 + rhalf + r;
            if (row < M) {
                float* cp = C + (size_t)row * ldc + n;
                if (accumulate) *cp += acc[nt][r];
                else            *cp  = acc[nt][r] + b;
            }
        }
    }
}

// ---------------------------------------------------------------------------
// conversion kernels
// ---------------------------------------------------------------------------
__global__ void cvt_f16_k(const float* __restrict__ in, _Float16* __restrict__ out, size_t n) {
    size_t i = (size_t)blockIdx.x * blockDim.x + threadIdx.x;
    if (i < n) out[i] = (_Float16)in[i];
}

__global__ void relu_cvt_k(const float* __restrict__ in, _Float16* __restrict__ out, size_t n) {
    size_t i = (size_t)blockIdx.x * blockDim.x + threadIdx.x;
    if (i < n) out[i] = (_Float16)fmaxf(in[i], 0.0f);
}

// ---------------------------------------------------------------------------
// conv1 (d=256) edge kernels: wave per edge, lane covers 8 channels
// ---------------------------------------------------------------------------
__global__ void logits256_k(const float* __restrict__ q, const float* __restrict__ k,
                            const int* __restrict__ src, const int* __restrict__ dst,
                            const float* __restrict__ ea, const float* __restrict__ We,
                            float* __restrict__ lg, int E) {
    int lane = threadIdx.x & 31;
    int e = blockIdx.x * (blockDim.x >> 5) + (threadIdx.x >> 5);
    if (e >= E) return;
    int s = src[e], d = dst[e];
    float eat[6];
#pragma unroll
    for (int t = 0; t < 6; ++t) eat[t] = ea[(size_t)e * 6 + t];
    const float* qp = q + (size_t)d * 256 + lane * 8;
    const float* kp = k + (size_t)s * 256 + lane * 8;
    const float* wp = We + (size_t)lane * 8 * 6;
    float acc = 0.f;
#pragma unroll
    for (int c = 0; c < 8; ++c) {
        float ec = 0.f;
#pragma unroll
        for (int t = 0; t < 6; ++t) ec += eat[t] * wp[c * 6 + t];
        acc += qp[c] * (kp[c] + ec);
    }
    acc = waveSum(acc);
    if (lane == 0) lg[e] = acc * (1.0f / 16.0f);   // 1/sqrt(256)
}

__global__ void init_ms_k(float* __restrict__ m, float* __restrict__ s, int n) {
    int i = blockIdx.x * blockDim.x + threadIdx.x;
    if (i < n) { m[i] = -INFINITY; s[i] = 0.0f; }
}

__global__ void segmax_k(const float* __restrict__ lg, const int* __restrict__ dst,
                         float* __restrict__ m, int E) {
    int e = blockIdx.x * blockDim.x + threadIdx.x;
    if (e < E) atomicMaxF(&m[dst[e]], lg[e]);
}

__global__ void expsum_k(float* __restrict__ lg, const int* __restrict__ dst,
                         const float* __restrict__ m, float* __restrict__ s, int E) {
    int e = blockIdx.x * blockDim.x + threadIdx.x;
    if (e >= E) return;
    int d = dst[e];
    float ex = __expf(lg[e] - m[d]);
    lg[e] = ex;
    atomicAdd(&s[d], ex);
}

__global__ void scatter256_k(const float* __restrict__ lg, const float* __restrict__ sden,
                             const float* __restrict__ v,
                             const int* __restrict__ src, const int* __restrict__ dst,
                             const float* __restrict__ ea, const float* __restrict__ We,
                             float* __restrict__ out, int E) {
    int lane = threadIdx.x & 31;
    int e = blockIdx.x * (blockDim.x >> 5) + (threadIdx.x >> 5);
    if (e >= E) return;
    int s = src[e], d = dst[e];
    float alpha = lg[e] / sden[d];
    float eat[6];
#pragma unroll
    for (int t = 0; t < 6; ++t) eat[t] = ea[(size_t)e * 6 + t];
    const float* vp = v + (size_t)s * 256 + lane * 8;
    const float* wp = We + (size_t)lane * 8 * 6;
    float* op = out + (size_t)d * 256 + lane * 8;
#pragma unroll
    for (int c = 0; c < 8; ++c) {
        float ec = 0.f;
#pragma unroll
        for (int t = 0; t < 6; ++t) ec += eat[t] * wp[c * 6 + t];
        atomicAdd(&op[c], alpha * (vp[c] + ec));
    }
}

// LayerNorm(256) + leaky_relu(0.01); reads f32, writes f16 (GEMM operand)
__global__ void ln_leaky256_k(const float* __restrict__ x, _Float16* __restrict__ outh, int n) {
    int lane = threadIdx.x & 31;
    int row = blockIdx.x * (blockDim.x >> 5) + (threadIdx.x >> 5);
    if (row >= n) return;
    const float* p = x + (size_t)row * 256 + lane * 8;
    _Float16* oh = outh + (size_t)row * 256 + lane * 8;
    float v[8], s = 0.f, s2 = 0.f;
#pragma unroll
    for (int c = 0; c < 8; ++c) { v[c] = p[c]; s += v[c]; s2 += v[c] * v[c]; }
    s = waveSum(s); s2 = waveSum(s2);
    float mu = s * (1.0f / 256.0f);
    float var = s2 * (1.0f / 256.0f) - mu * mu;
    float inv = rsqrtf(var + 1e-5f);
#pragma unroll
    for (int c = 0; c < 8; ++c) {
        float y = (v[c] - mu) * inv;
        oh[c] = (_Float16)((y > 0.f) ? y : 0.01f * y);
    }
}

// ---------------------------------------------------------------------------
// small convs (cin=10 -> d=20) and tail layers
// ---------------------------------------------------------------------------
__global__ void qkvs_small_k(const float* __restrict__ x,
                             const float* __restrict__ Wq, const float* __restrict__ bq,
                             const float* __restrict__ Wk, const float* __restrict__ bk,
                             const float* __restrict__ Wv, const float* __restrict__ bv,
                             const float* __restrict__ Ws, const float* __restrict__ bs,
                             float* __restrict__ q, float* __restrict__ k,
                             float* __restrict__ v, float* __restrict__ o, int n) {
    int i = blockIdx.x * blockDim.x + threadIdx.x;
    if (i >= n) return;
    float xi[10];
#pragma unroll
    for (int c = 0; c < 10; ++c) xi[c] = x[(size_t)i * 10 + c];
#pragma unroll
    for (int j = 0; j < 20; ++j) {
        float aq = bq[j], ak = bk[j], av = bv[j], as = bs[j];
#pragma unroll
        for (int c = 0; c < 10; ++c) {
            aq += Wq[j * 10 + c] * xi[c];
            ak += Wk[j * 10 + c] * xi[c];
            av += Wv[j * 10 + c] * xi[c];
            as += Ws[j * 10 + c] * xi[c];
        }
        q[(size_t)i * 20 + j] = aq;
        k[(size_t)i * 20 + j] = ak;
        v[(size_t)i * 20 + j] = av;
        o[(size_t)i * 20 + j] = as;   // skip/root term, scatter accumulates on top
    }
}

__global__ void logits20_k(const float* __restrict__ q, const float* __restrict__ k,
                           const int* __restrict__ src, const int* __restrict__ dst,
                           const float* __restrict__ ea, const float* __restrict__ We,
                           float* __restrict__ lg, int E) {
    int e = blockIdx.x * blockDim.x + threadIdx.x;
    if (e >= E) return;
    int s = src[e], d = dst[e];
    float eat[6];
#pragma unroll
    for (int t = 0; t < 6; ++t) eat[t] = ea[(size_t)e * 6 + t];
    float acc = 0.f;
#pragma unroll
    for (int c = 0; c < 20; ++c) {
        float ec = 0.f;
#pragma unroll
        for (int t = 0; t < 6; ++t) ec += eat[t] * We[c * 6 + t];
        acc += q[(size_t)d * 20 + c] * (k[(size_t)s * 20 + c] + ec);
    }
    lg[e] = acc * 0.22360679775f;   // 1/sqrt(20)
}

__global__ void scatter20_k(const float* __restrict__ lg, const float* __restrict__ sden,
                            const float* __restrict__ v,
                            const int* __restrict__ src, const int* __restrict__ dst,
                            const float* __restrict__ ea, const float* __restrict__ We,
                            float* __restrict__ out, int E) {
    int e = blockIdx.x * blockDim.x + threadIdx.x;
    if (e >= E) return;
    int s = src[e], d = dst[e];
    float alpha = lg[e] / sden[d];
    float eat[6];
#pragma unroll
    for (int t = 0; t < 6; ++t) eat[t] = ea[(size_t)e * 6 + t];
#pragma unroll
    for (int c = 0; c < 20; ++c) {
        float ec = 0.f;
#pragma unroll
        for (int t = 0; t < 6; ++t) ec += eat[t] * We[c * 6 + t];
        atomicAdd(&out[(size_t)d * 20 + c], alpha * (v[(size_t)s * 20 + c] + ec));
    }
}

__global__ void ln_relu20_k(float* __restrict__ x, int n) {
    int i = blockIdx.x * blockDim.x + threadIdx.x;
    if (i >= n) return;
    float* p = x + (size_t)i * 20;
    float v[20], s = 0.f, s2 = 0.f;
#pragma unroll
    for (int c = 0; c < 20; ++c) { v[c] = p[c]; s += v[c]; s2 += v[c] * v[c]; }
    float mu = s * 0.05f;
    float var = s2 * 0.05f - mu * mu;
    float inv = rsqrtf(var + 1e-5f);
#pragma unroll
    for (int c = 0; c < 20; ++c) {
        float y = (v[c] - mu) * inv;
        p[c] = fmaxf(y, 0.f);
    }
}

__global__ void ln_leaky10_k(float* __restrict__ x, int n) {
    int i = blockIdx.x * blockDim.x + threadIdx.x;
    if (i >= n) return;
    float* p = x + (size_t)i * 10;
    float v[10], s = 0.f, s2 = 0.f;
#pragma unroll
    for (int c = 0; c < 10; ++c) { v[c] = p[c]; s += v[c]; s2 += v[c] * v[c]; }
    float mu = s * 0.1f;
    float var = s2 * 0.1f - mu * mu;
    float inv = rsqrtf(var + 1e-5f);
#pragma unroll
    for (int c = 0; c < 10; ++c) {
        float y = (v[c] - mu) * inv;
        p[c] = (y > 0.f) ? y : 0.01f * y;
    }
}

__global__ void lin4_ln_leaky_k(const float* __restrict__ o2, const float* __restrict__ z,
                                const float* __restrict__ W4, const float* __restrict__ b4,
                                float* __restrict__ z4, int n) {
    int i = blockIdx.x * blockDim.x + threadIdx.x;
    if (i >= n) return;
    const float* oc = o2 + (size_t)i * 20;
    const float* zc = z + (size_t)i * 10;
    float y[10], s = 0.f, s2 = 0.f;
#pragma unroll
    for (int j = 0; j < 10; ++j) {
        float a = b4[j];
#pragma unroll
        for (int c = 0; c < 20; ++c) a += W4[j * 30 + c] * oc[c];
#pragma unroll
        for (int c = 0; c < 10; ++c) a += W4[j * 30 + 20 + c] * zc[c];
        y[j] = a; s += a; s2 += a * a;
    }
    float mu = s * 0.1f;
    float var = s2 * 0.1f - mu * mu;
    float inv = rsqrtf(var + 1e-5f);
#pragma unroll
    for (int j = 0; j < 10; ++j) {
        float t = (y[j] - mu) * inv;
        z4[(size_t)i * 10 + j] = (t > 0.f) ? t : 0.01f * t;
    }
}

__global__ void lin5_softplus_k(const float* __restrict__ o3, const float* __restrict__ z4,
                                const float* __restrict__ W5, const float* __restrict__ b5,
                                float* __restrict__ conc, int n) {
    int i = blockIdx.x * blockDim.x + threadIdx.x;
    if (i >= n) return;
    float a = b5[0];
#pragma unroll
    for (int c = 0; c < 20; ++c) a += W5[c] * o3[(size_t)i * 20 + c];
#pragma unroll
    for (int c = 0; c < 10; ++c) a += W5[20 + c] * z4[(size_t)i * 10 + c];
    conc[i] = (a > 30.f) ? a : log1pf(__expf(a));
}

__global__ void final_norm_k(const float* __restrict__ conc, float* __restrict__ out, int G) {
    int g = blockIdx.x * blockDim.x + threadIdx.x;
    if (g >= G) return;
    float s = 1e-20f;
#pragma unroll
    for (int a = 0; a < 6; ++a) s += conc[g * 6 + a];
#pragma unroll
    for (int a = 0; a < 6; ++a) out[g * 6 + a] = conc[g * 6 + a] / s;
}

// ---------------------------------------------------------------------------
// host side
// ---------------------------------------------------------------------------
static inline void gemm256(const _Float16* A, int lda, const _Float16* W, int ldw,
                           const float* bias, float* C, int ldc,
                           int M, int K, int acc, hipStream_t st) {
    // N == 256, wave tile 16x128 -> grid.x = 2
    dim3 g(2, (M + 63) / 64);
    size_t smem = (size_t)128 * (K + 8) * sizeof(_Float16);
    gemm_h<8, false><<<g, dim3(128), smem, st>>>(A, lda, W, ldw, bias, C, ldc, M, 256, K, acc);
}

static inline void cvt(const float* in, _Float16* out, size_t n, hipStream_t st) {
    cvt_f16_k<<<(int)((n + 255) / 256), 256, 0, st>>>(in, out, n);
}

static void run_small_conv(const float* x,
                           const float* Wq, const float* bq, const float* Wk, const float* bk,
                           const float* Wv, const float* bv, const float* We,
                           const float* Ws, const float* bs,
                           const int* src, const int* dst, const float* ea,
                           float* q, float* k, float* v, float* out,
                           float* lg, float* m, float* sden, hipStream_t st) {
    int nb = (N_NODES + 255) / 256;
    int eb = (N_EDGES + 255) / 256;
    qkvs_small_k<<<nb, 256, 0, st>>>(x, Wq, bq, Wk, bk, Wv, bv, Ws, bs, q, k, v, out, N_NODES);
    init_ms_k<<<nb, 256, 0, st>>>(m, sden, N_NODES);
    logits20_k<<<eb, 256, 0, st>>>(q, k, src, dst, ea, We, lg, N_EDGES);
    segmax_k<<<eb, 256, 0, st>>>(lg, dst, m, N_EDGES);
    expsum_k<<<eb, 256, 0, st>>>(lg, dst, m, sden, N_EDGES);
    scatter20_k<<<eb, 256, 0, st>>>(lg, sden, v, src, dst, ea, We, out, N_EDGES);
    ln_relu20_k<<<nb, 256, 0, st>>>(out, N_NODES);
}

extern "C" void kernel_launch(void* const* d_in, const int* in_sizes, int n_in,
                              void* d_out, int out_size, void* d_ws, size_t ws_size,
                              hipStream_t stream) {
    (void)in_sizes; (void)n_in; (void)out_size; (void)ws_size;

    const float* state = (const float*)d_in[0];
    const int*   ei    = (const int*)d_in[1];
    const float* ea    = (const float*)d_in[2];
    const int* src = ei;
    const int* dst = ei + N_EDGES;

    const float *Wq1 = (const float*)d_in[3],  *bq1 = (const float*)d_in[4];
    const float *Wk1 = (const float*)d_in[5],  *bk1 = (const float*)d_in[6];
    const float *Wv1 = (const float*)d_in[7],  *bv1 = (const float*)d_in[8];
    const float *We1 = (const float*)d_in[9];
    const float *Ws1 = (const float*)d_in[10], *bs1 = (const float*)d_in[11];
    const float *W1 = (const float*)d_in[12], *b1 = (const float*)d_in[13];
    const float *W2 = (const float*)d_in[14], *b2 = (const float*)d_in[15];
    const float *W3 = (const float*)d_in[16], *b3 = (const float*)d_in[17];
    const float *Wq2 = (const float*)d_in[18], *bq2 = (const float*)d_in[19];
    const float *Wk2 = (const float*)d_in[20], *bk2 = (const float*)d_in[21];
    const float *Wv2 = (const float*)d_in[22], *bv2 = (const float*)d_in[23];
    const float *We2 = (const float*)d_in[24];
    const float *Ws2 = (const float*)d_in[25], *bs2 = (const float*)d_in[26];
    const float *W4 = (const float*)d_in[27], *b4 = (const float*)d_in[28];
    const float *Wq3 = (const float*)d_in[29], *bq3 = (const float*)d_in[30];
    const float *Wk3 = (const float*)d_in[31], *bk3 = (const float*)d_in[32];
    const float *Wv3 = (const float*)d_in[33], *bv3 = (const float*)d_in[34];
    const float *We3 = (const float*)d_in[35];
    const float *Ws3 = (const float*)d_in[36], *bs3 = (const float*)d_in[37];
    const float *W5 = (const float*)d_in[38], *b5 = (const float*)d_in[39];

    float* out = (float*)d_out;

    // ---------------- workspace carve-up ----------------
    float* ws = (float*)d_ws;
    size_t off = 0;
    float* q1 = ws + off; off += (size_t)N_NODES * 256;
    float* k1 = ws + off; off += (size_t)N_NODES * 256;
    float* v1 = ws + off; off += (size_t)N_NODES * 256;   // f32 until scatter; then f16 slots
    float* o1 = ws + off; off += (size_t)N_NODES * 256;
    float* lg = ws + off; off += N_EDGES;
    float* mN = ws + off; off += N_NODES;
    float* sN = ws + off; off += N_NODES;
    float* z  = ws + off; off += (size_t)N_NODES * 10;
    float* q2 = ws + off; off += (size_t)N_NODES * 20;
    float* k2 = ws + off; off += (size_t)N_NODES * 20;
    float* v2 = ws + off; off += (size_t)N_NODES * 20;
    float* o2 = ws + off; off += (size_t)N_NODES * 20;
    float* z4 = ws + off; off += (size_t)N_NODES * 10;
    float* cc = ws + off; off += N_NODES;
    // f16 scratch (fresh region): state + weights
    _Float16* stateH = (_Float16*)(ws + off); off += (size_t)N_NODES * 128 / 2;
    _Float16* Wq1H = (_Float16*)(ws + off); off += (256 * 128) / 2;
    _Float16* Wk1H = (_Float16*)(ws + off); off += (256 * 128) / 2;
    _Float16* Wv1H = (_Float16*)(ws + off); off += (256 * 128) / 2;
    _Float16* Ws1H = (_Float16*)(ws + off); off += (256 * 128) / 2;
    _Float16* W1H  = (_Float16*)(ws + off); off += (256 * 384) / 2;
    _Float16* W2H  = (_Float16*)(ws + off); off += (256 * 256) / 2;
    _Float16* W3H  = (_Float16*)(ws + off); off += (10 * 256) / 2 + 16;
    // activation f16 buffers overlay the retired v1 region (disjoint lifetimes)
    _Float16* o1H = (_Float16*)v1;                        // live: relu_cvt -> lin1a
    _Float16* y1H = o1H + (size_t)N_NODES * 256;          // live: ln(y1) -> lin2
    _Float16* y2H = o1H;                                  // live: ln(y2) -> lin3

    int nb  = (N_NODES + 255) / 256;
    int eb  = (N_EDGES + 255) / 256;
    int ewb = (N_EDGES + 7) / 8;
    int nwb = (N_NODES + 7) / 8;

    // ---------------- f16 pre-conversion ----------------
    cvt(state, stateH, (size_t)N_NODES * 128, stream);
    cvt(Wq1, Wq1H, 256 * 128, stream);
    cvt(Wk1, Wk1H, 256 * 128, stream);
    cvt(Wv1, Wv1H, 256 * 128, stream);
    cvt(Ws1, Ws1H, 256 * 128, stream);
    cvt(W1, W1H, 256 * 384, stream);
    cvt(W2, W2H, 256 * 256, stream);
    cvt(W3, W3H, 10 * 256, stream);

    // ---------------- conv1 (128 -> 256), WMMA ----------------
    gemm256(stateH, 128, Wq1H, 128, bq1, q1, 256, N_NODES, 128, 0, stream);
    gemm256(stateH, 128, Wk1H, 128, bk1, k1, 256, N_NODES, 128, 0, stream);
    gemm256(stateH, 128, Wv1H, 128, bv1, v1, 256, N_NODES, 128, 0, stream);
    gemm256(stateH, 128, Ws1H, 128, bs1, o1, 256, N_NODES, 128, 0, stream); // skip term

    init_ms_k<<<nb, 256, 0, stream>>>(mN, sN, N_NODES);
    logits256_k<<<ewb, 256, 0, stream>>>(q1, k1, src, dst, ea, We1, lg, N_EDGES);
    segmax_k<<<eb, 256, 0, stream>>>(lg, dst, mN, N_EDGES);
    expsum_k<<<eb, 256, 0, stream>>>(lg, dst, mN, sN, N_EDGES);
    scatter256_k<<<ewb, 256, 0, stream>>>(lg, sN, v1, src, dst, ea, We1, o1, N_EDGES);
    // relu + f16 convert (v1 region is free now)
    relu_cvt_k<<<(int)(((size_t)N_NODES * 256 + 255) / 256), 256, 0, stream>>>(
        o1, o1H, (size_t)N_NODES * 256);

    // ---------------- MLP head (lin1, lin2, lin3), WMMA ----------------
    float* y1 = q1;   // reuse f32 regions for GEMM outputs
    float* y2 = k1;
    gemm256(o1H, 256, W1H, 384, b1, y1, 256, N_NODES, 256, 0, stream);          // lin1 (out part)
    gemm256(stateH, 128, W1H + 256, 384, nullptr, y1, 256, N_NODES, 128, 1, stream); // + state part
    ln_leaky256_k<<<nwb, 256, 0, stream>>>(y1, y1H, N_NODES);
    gemm256(y1H, 256, W2H, 256, b2, y2, 256, N_NODES, 256, 0, stream);          // lin2
    ln_leaky256_k<<<nwb, 256, 0, stream>>>(y2, y2H, N_NODES);
    {   // lin3 -> z[60000 x 10] (guarded, single 16-col tile)
        dim3 g(1, (N_NODES + 63) / 64);
        size_t smem = (size_t)16 * (256 + 8) * sizeof(_Float16);
        gemm_h<1, true><<<g, dim3(128), smem, stream>>>(
            y2H, 256, W3H, 256, b3, z, 10, N_NODES, 10, 256, 0);
    }
    ln_leaky10_k<<<nb, 256, 0, stream>>>(z, N_NODES);

    // ---------------- conv2 (10 -> 20) + lin4 ----------------
    run_small_conv(z, Wq2, bq2, Wk2, bk2, Wv2, bv2, We2, Ws2, bs2,
                   src, dst, ea, q2, k2, v2, o2, lg, mN, sN, stream);
    lin4_ln_leaky_k<<<nb, 256, 0, stream>>>(o2, z, W4, b4, z4, N_NODES);

    // ---------------- conv3 (10 -> 20) + lin5 ----------------
    run_small_conv(z4, Wq3, bq3, Wk3, bk3, Wv3, bv3, We3, Ws3, bs3,
                   src, dst, ea, q2, k2, v2, o2, lg, mN, sN, stream);
    lin5_softplus_k<<<nb, 256, 0, stream>>>(o2, z4, W5, b5, cc, N_NODES);

    // ---------------- final per-graph normalization ----------------
    final_norm_k<<<(G_NUM + 255) / 256, 256, 0, stream>>>(cc, out, G_NUM);
}
